// SelfAttention_34454227648833
// MI455X (gfx1250) — matrix operand order repaired and checked by reading, not compiled, not run on hardware
//
#include <hip/hip_runtime.h>
#include <cstdint>
#include <cstddef>

// ---------------------------------------------------------------------------
// MI455X (gfx1250) self-attention block, bf16 WMMA everywhere.
// H=32, HKV=8, G=4, D=64, B=2, T=2048, C=2048.
// Round 2: stage A-tiles / K-chunks with GLOBAL_LOAD_ASYNC_TO_LDS_B128
// (ASYNCcnt path) instead of global->VGPR->ds_store.
// ---------------------------------------------------------------------------

typedef __attribute__((ext_vector_type(16))) __bf16 v16bf;
typedef __attribute__((ext_vector_type(8)))  float  v8f;

#define B_    2
#define T_    2048
#define C_    2048
#define H_    32
#define HKV_  8
#define G_    4
#define D_    64

// ---------------------------------------------------------------------------
// CDNA5 async copy: cache/memory -> LDS, tracked by ASYNCcnt.
// dsaddr = low 32 bits of the generic shared pointer (LDS aperture strips
// the upper bits, ISA 10.2); vaddr = 64-bit global address in a VGPR pair.
// ---------------------------------------------------------------------------
__device__ __forceinline__ void async_load_lds_b128(void* lds, const void* g) {
  uint32_t l = (uint32_t)(uintptr_t)lds;
  unsigned long long a = (unsigned long long)(uintptr_t)g;
  asm volatile("global_load_async_to_lds_b128 %0, %1, off"
               :: "v"(l), "v"(a) : "memory");
}
__device__ __forceinline__ void wait_async() {
  asm volatile("s_wait_asynccnt 0" ::: "memory");
}

// ---------------------------------------------------------------------------
// WMMA wrapper: D(16x16 f32) = A(16x32 bf16) * B(32x16 bf16) + C
// ---------------------------------------------------------------------------
__device__ __forceinline__ v8f wmma_bf16(v16bf a, v16bf b, v8f c) {
  return __builtin_amdgcn_wmma_f32_16x16x32_bf16(
      /*neg_a=*/false, a, /*neg_b=*/false, b,
      /*c_mod=*/(short)0, c, /*reuse_a=*/false, /*reuse_b=*/false);
}

// A-fragment (16x32 bf16) from LDS, row-major [row][k], stride `ld` elems.
__device__ __forceinline__ v16bf lds_afrag(const __bf16* base, int ld,
                                           int rowBase, int lane) {
  const int row = rowBase + (lane & 15);
  const int kg  = (lane >> 4) << 3;  // +8 K for lanes 16..31
  const uint32_t* p = (const uint32_t*)(base + row * ld);
  v16bf a;
#pragma unroll
  for (int i = 0; i < 8; ++i) {
    const int kk = ((i & 3) << 1) + ((i >> 2) << 4) + kg;  // even
    union { uint32_t u; __bf16 h[2]; } cv;
    cv.u = p[kk >> 1];
    a[2 * i]     = cv.h[0];
    a[2 * i + 1] = cv.h[1];
  }
  return a;
}

// B-fragment (32x16 bf16) from LDS stored K-contiguous: [col][k], stride `ld`.
__device__ __forceinline__ v16bf lds_bfrag(const __bf16* base, int ld,
                                           int colBase, int lane) {
  const int col  = colBase + (lane & 15);
  const int koff = (lane >> 4) << 3;  // +8 K for lanes 16..31
  const uint32_t* p = (const uint32_t*)(base + col * ld);
  v16bf b;
#pragma unroll
  for (int i = 0; i < 8; ++i) {
    const int kk = ((i & 3) << 1) + ((i >> 2) << 4) + koff;
    union { uint32_t u; __bf16 h[2]; } cv;
    cv.u = p[kk >> 1];
    b[2 * i]     = cv.h[0];
    b[2 * i + 1] = cv.h[1];
  }
  return b;
}

// ---------------------------------------------------------------------------
// fp32 -> bf16 conversion (4 elems/thread, b128 load, b64 store)
// ---------------------------------------------------------------------------
__global__ __launch_bounds__(256)
void cvt_f32_bf16(const float* __restrict__ in, __bf16* __restrict__ out, int n4) {
  int i = blockIdx.x * 256 + threadIdx.x;
  if (i >= n4) return;
  float4 f = ((const float4*)in)[i];
  union { __bf16 h[4]; uint64_t u; } cv;
  cv.h[0] = (__bf16)f.x; cv.h[1] = (__bf16)f.y;
  cv.h[2] = (__bf16)f.z; cv.h[3] = (__bf16)f.w;
  ((uint64_t*)out)[i] = cv.u;
}

// ---------------------------------------------------------------------------
// Tiled bf16 GEMM: C(MxN) = A(MxK) * B(KxN). 256 threads = 8 waves.
// Block tile 128x64, wave tile 32x32 (2x2 of 16x16 WMMA), BK=64.
// A tile staged via async LDS copies; B staged transposed via VGPRs.
// ---------------------------------------------------------------------------
template <typename TOUT>
__global__ __launch_bounds__(256)
void gemm_bf16(const __bf16* __restrict__ A, const __bf16* __restrict__ Bm,
               TOUT* __restrict__ Cm, int M, int N, int K) {
  __shared__ __bf16 As[128][72];  // [m][k], 144B row stride (16B aligned)
  __shared__ __bf16 Bt[64][72];   // [n][k], K-contiguous for B fragments

  const int tid  = threadIdx.x;
  const int lane = tid & 31;
  const int w    = tid >> 5;
  const int wm   = (w >> 1) * 32;  // 4 waves along M
  const int wn   = (w & 1) * 32;   // 2 waves along N
  const int m0   = blockIdx.y * 128;
  const int n0   = blockIdx.x * 64;

  v8f z = {};
  v8f acc[2][2];
  acc[0][0] = z; acc[0][1] = z; acc[1][0] = z; acc[1][1] = z;

  // A staging: 2 threads per row, 32 bf16 (64B) each -> 4 async b128
  const int arow = tid >> 1;
  const int acol = (tid & 1) << 5;
  // B staging: 4 threads per K-row, 16 bf16 (32B) each, scattered into Bt
  const int brow = tid >> 2;
  const int bcol = (tid & 3) << 4;

  const __bf16* aG = A  + (size_t)(m0 + arow) * K + acol;
  const __bf16* bG = Bm + (size_t)brow * N + n0 + bcol;

  for (int k0 = 0; k0 < K; k0 += 64) {
    // A tile: direct cache->LDS async copies (no VGPR round trip)
#pragma unroll
    for (int j = 0; j < 4; ++j)
      async_load_lds_b128(&As[arow][acol + 8 * j], aG + k0 + 8 * j);

    {  // B tile -> LDS (transposed through VGPRs)
      union { uint4 v[2]; __bf16 h[16]; } bb;
      const uint4* src = (const uint4*)(bG + (size_t)k0 * N);
      bb.v[0] = src[0];
      bb.v[1] = src[1];
#pragma unroll
      for (int j = 0; j < 16; ++j) Bt[bcol + j][brow] = bb.h[j];
    }
    if (k0 + 64 < K) {  // hint next K-tile into cache (global_prefetch_b8)
      __builtin_prefetch(aG + k0 + 64, 0, 1);
      __builtin_prefetch(bG + (size_t)(k0 + 64) * N, 0, 1);
    }
    wait_async();        // ASYNCcnt == 0: our LDS writes have landed
    __syncthreads();

#pragma unroll
    for (int ks = 0; ks < 2; ++ks) {
      v16bf af0 = lds_afrag(&As[0][0] + ks * 32, 72, wm + 0,  lane);
      v16bf af1 = lds_afrag(&As[0][0] + ks * 32, 72, wm + 16, lane);
      v16bf bf0 = lds_bfrag(&Bt[0][0] + ks * 32, 72, wn + 0,  lane);
      v16bf bf1 = lds_bfrag(&Bt[0][0] + ks * 32, 72, wn + 16, lane);
      acc[0][0] = wmma_bf16(af0, bf0, acc[0][0]);
      acc[0][1] = wmma_bf16(af0, bf1, acc[0][1]);
      acc[1][0] = wmma_bf16(af1, bf0, acc[1][0]);
      acc[1][1] = wmma_bf16(af1, bf1, acc[1][1]);
    }
    __syncthreads();
  }

  const int laneN = lane & 15;
  const int kg    = lane >> 4;
#pragma unroll
  for (int ms = 0; ms < 2; ++ms)
#pragma unroll
    for (int ns = 0; ns < 2; ++ns)
#pragma unroll
      for (int r = 0; r < 8; ++r) {
        int m = m0 + wm + ms * 16 + r + kg * 8;
        int n = n0 + wn + ns * 16 + laneN;
        Cm[(size_t)m * N + n] = (TOUT)acc[ms][ns][r];
      }
}

// ---------------------------------------------------------------------------
// RoPE + transpose to head-major: in [B*T, nheads*D] -> out [B, nheads, T, D]
// ---------------------------------------------------------------------------
__global__ __launch_bounds__(256)
void rope_kernel(const __bf16* __restrict__ in_tok, __bf16* __restrict__ out_head,
                 int nheads) {
  int idx = blockIdx.x * 256 + threadIdx.x;
  int i  = idx & 31;
  int hh = (idx >> 5) % nheads;
  int t  = (idx / (32 * nheads)) % T_;
  int b  = idx / (32 * nheads * T_);
  const __bf16* src = in_tok + (size_t)(b * T_ + t) * (nheads * D_) + hh * D_;
  float x1 = (float)src[i];
  float x2 = (float)src[i + 32];
  float freq = __powf(10000.0f, -(float)(2 * i) / 64.0f);
  float ang  = (float)t * freq;
  float sn, cs;
  __sincosf(ang, &sn, &cs);
  __bf16* dst = out_head + ((size_t)(b * nheads + hh) * T_ + t) * D_;
  dst[i]      = (__bf16)(x1 * cs - x2 * sn);
  dst[i + 32] = (__bf16)(x2 * cs + x1 * sn);
}

// V: token-major [B*T, HKV*D] -> head-major [B, HKV, T, D]
__global__ __launch_bounds__(256)
void v_transpose(const __bf16* __restrict__ in_tok, __bf16* __restrict__ out_head) {
  int idx = blockIdx.x * 256 + threadIdx.x;  // B*T*HKV*D = 2^21
  int d  = idx & 63;
  int hh = (idx >> 6) & 7;
  int t  = (idx >> 9) & 2047;
  int b  = idx >> 20;
  out_head[((size_t)(b * HKV_ + hh) * T_ + t) * D_ + d] =
      in_tok[(size_t)(b * T_ + t) * (HKV_ * D_) + hh * D_ + d];
}

// ---------------------------------------------------------------------------
// Flash attention (causal, GQA). Block = 128 threads = 4 waves.
// K chunk staged via async LDS copies; V transposed via VGPRs.
// ---------------------------------------------------------------------------
__global__ __launch_bounds__(128)
void attn_kernel(const __bf16* __restrict__ qh, const __bf16* __restrict__ kh,
                 const __bf16* __restrict__ vh, __bf16* __restrict__ attn_tok) {
  const int head = blockIdx.x;      // b*H + h
  const int b    = head >> 5;
  const int h    = head & 31;
  const int hk   = h >> 2;          // G = 4
  const int t0b  = blockIdx.y * 64;
  const int tid  = threadIdx.x;
  const int lane = tid & 31;
  const int w    = tid >> 5;
  const int t0w  = t0b + w * 16;
  const int laneN = lane & 15;
  const int kg    = lane >> 4;

  __shared__ __bf16 Ks[32][72];     // [s][d], 144B row stride (16B aligned)
  __shared__ __bf16 Vt[64][40];     // [d][s]   (V chunk, transposed)
  __shared__ __bf16 Pw[4][16][40];  // per-wave P tile [m][s]

  const __bf16* qbase = qh + (size_t)(b * H_ + h) * T_ * D_;
  const __bf16* kbase = kh + (size_t)(b * HKV_ + hk) * T_ * D_;
  const __bf16* vbase = vh + (size_t)(b * HKV_ + hk) * T_ * D_;

  // Q fragments (16x64, split as two 16x32 A-fragments), loaded once.
  v16bf qf[2];
  {
    const __bf16* qrow = qbase + (size_t)(t0w + laneN) * D_;
#pragma unroll
    for (int half = 0; half < 2; ++half) {
      v16bf a;
#pragma unroll
      for (int i = 0; i < 8; ++i) {
        const int kk = ((i & 3) << 1) + ((i >> 2) << 4) + (kg << 3) + half * 32;
        union { uint32_t u; __bf16 h2[2]; } cv;
        cv.u = *(const uint32_t*)(qrow + kk);
        a[2 * i]     = cv.h2[0];
        a[2 * i + 1] = cv.h2[1];
      }
      qf[half] = a;
    }
  }

  v8f z = {};
  v8f accO[4];  // D split into 4 groups of 16
  float mst[8], lst[8];
#pragma unroll
  for (int g = 0; g < 4; ++g) accO[g] = z;
#pragma unroll
  for (int r = 0; r < 8; ++r) { mst[r] = -3.0e38f; lst[r] = 0.0f; }

  const int nChunks = (t0b + 64) / 32;  // causal: keys up to t0b+63
  for (int c = 0; c < nChunks; ++c) {
    const int s0 = c * 32;

    {  // stage K (async cache->LDS) and V (VGPR transpose) chunk
      const int row  = tid >> 2;        // 0..31
      const int colb = (tid & 3) << 4;  // 0,16,32,48
      const __bf16* kr = kbase + (size_t)(s0 + row) * D_ + colb;
      async_load_lds_b128(&Ks[row][colb + 0], kr + 0);
      async_load_lds_b128(&Ks[row][colb + 8], kr + 8);

      const __bf16* vr = vbase + (size_t)(s0 + row) * D_ + colb;
      union { uint4 v[2]; __bf16 h[16]; } vv;
      vv.v[0] = *(const uint4*)(vr);
      vv.v[1] = *(const uint4*)(vr + 8);
#pragma unroll
      for (int j = 0; j < 16; ++j) Vt[colb + j][row] = vv.h[j];
    }
    wait_async();
    __syncthreads();

    if (s0 <= t0w + 15) {  // wave-uniform: this chunk intersects causal range
      // ---- scores S = (Q K^T) * scale, two 16x16 tiles over 32 keys ----
      v8f sv[2];
#pragma unroll
      for (int j = 0; j < 2; ++j) {
        v8f acc = z;
#pragma unroll
        for (int half = 0; half < 2; ++half) {
          v16bf bk = lds_bfrag(&Ks[0][0] + half * 32, 72, j * 16, lane);
          acc = wmma_bf16(qf[half], bk, acc);
        }
        sv[j] = acc;
      }

      // ---- scale + causal mask + row max (butterfly over 16-lane half) ----
      const float scale = 0.125f;  // 1/sqrt(64)
      float rmax[8];
#pragma unroll
      for (int r = 0; r < 8; ++r) {
        const int qr = t0w + r + kg * 8;
#pragma unroll
        for (int j = 0; j < 2; ++j) {
          float s = sv[j][r] * scale;
          int col = s0 + j * 16 + laneN;
          sv[j][r] = (col <= qr) ? s : -3.0e38f;
        }
        float mx = fmaxf(sv[0][r], sv[1][r]);
        mx = fmaxf(mx, __shfl_xor(mx, 8, 16));
        mx = fmaxf(mx, __shfl_xor(mx, 4, 16));
        mx = fmaxf(mx, __shfl_xor(mx, 2, 16));
        mx = fmaxf(mx, __shfl_xor(mx, 1, 16));
        rmax[r] = mx;
      }

      // ---- online softmax update + write P(bf16) into per-wave LDS ----
#pragma unroll
      for (int r = 0; r < 8; ++r) {
        float mnew = fmaxf(mst[r], rmax[r]);
        float corr = __expf(mst[r] - mnew);
        mst[r] = mnew;
        float psum = 0.0f;
#pragma unroll
        for (int j = 0; j < 2; ++j) {
          float p = __expf(sv[j][r] - mnew);
          sv[j][r] = p;
          psum += p;
        }
        psum += __shfl_xor(psum, 8, 16);
        psum += __shfl_xor(psum, 4, 16);
        psum += __shfl_xor(psum, 2, 16);
        psum += __shfl_xor(psum, 1, 16);
        lst[r] = lst[r] * corr + psum;
#pragma unroll
        for (int g = 0; g < 4; ++g) accO[g][r] *= corr;
#pragma unroll
        for (int j = 0; j < 2; ++j)
          Pw[w][r + kg * 8][j * 16 + laneN] = (__bf16)sv[j][r];
      }

      // ---- O += P(16x32) * V(32x64), 4 WMMAs over D groups ----
      v16bf pf = lds_afrag(&Pw[w][0][0], 40, 0, lane);
#pragma unroll
      for (int g = 0; g < 4; ++g) {
        v16bf bv = lds_bfrag(&Vt[0][0], 40, g * 16, lane);
        accO[g] = wmma_bf16(pf, bv, accO[g]);
      }
    }
    __syncthreads();
  }

  // ---- normalize and store token-major bf16 ----
  float inv[8];
#pragma unroll
  for (int r = 0; r < 8; ++r) inv[r] = 1.0f / lst[r];
#pragma unroll
  for (int g = 0; g < 4; ++g)
#pragma unroll
    for (int r = 0; r < 8; ++r) {
      int t = t0w + r + kg * 8;
      int d = g * 16 + laneN;
      attn_tok[(size_t)(b * T_ + t) * C_ + h * D_ + d] =
          (__bf16)(accO[g][r] * inv[r]);
    }
}

// ---------------------------------------------------------------------------
// Host launcher
// ---------------------------------------------------------------------------
extern "C" void kernel_launch(void* const* d_in, const int* in_sizes, int n_in,
                              void* d_out, int out_size, void* d_ws, size_t ws_size,
                              hipStream_t stream) {
  (void)in_sizes; (void)n_in; (void)out_size; (void)ws_size;
  const float* x  = (const float*)d_in[0];
  const float* Wq = (const float*)d_in[1];
  const float* Wk = (const float*)d_in[2];
  const float* Wv = (const float*)d_in[3];
  const float* Wo = (const float*)d_in[4];
  float* out = (float*)d_out;

  char* ws = (char*)d_ws;
  size_t off = 0;
  auto alloc = [&](size_t bytes) -> void* {
    void* p = ws + off;
    off += (bytes + 255) & ~(size_t)255;
    return p;
  };
  const size_t BT = (size_t)B_ * T_;
  __bf16* xb   = (__bf16*)alloc(BT * C_ * 2);
  __bf16* Wqb  = (__bf16*)alloc((size_t)C_ * H_ * D_ * 2);
  __bf16* Wkb  = (__bf16*)alloc((size_t)C_ * HKV_ * D_ * 2);
  __bf16* Wvb  = (__bf16*)alloc((size_t)C_ * HKV_ * D_ * 2);
  __bf16* Wob  = (__bf16*)alloc((size_t)C_ * C_ * 2);
  __bf16* qtok = (__bf16*)alloc(BT * H_ * D_ * 2);
  __bf16* ktok = (__bf16*)alloc(BT * HKV_ * D_ * 2);
  __bf16* vtok = (__bf16*)alloc(BT * HKV_ * D_ * 2);
  __bf16* qhd  = (__bf16*)alloc(BT * H_ * D_ * 2);
  __bf16* khd  = (__bf16*)alloc(BT * HKV_ * D_ * 2);
  __bf16* vhd  = (__bf16*)alloc(BT * HKV_ * D_ * 2);
  __bf16* attn = (__bf16*)alloc(BT * C_ * 2);

  auto cvt = [&](const float* src, __bf16* dst, size_t n) {
    cvt_f32_bf16<<<(unsigned)(n / 1024), 256, 0, stream>>>(src, dst, (int)(n / 4));
  };
  cvt(x,  xb,  BT * C_);
  cvt(Wq, Wqb, (size_t)C_ * H_ * D_);
  cvt(Wk, Wkb, (size_t)C_ * HKV_ * D_);
  cvt(Wv, Wvb, (size_t)C_ * HKV_ * D_);
  cvt(Wo, Wob, (size_t)C_ * C_);

  dim3 blk(256);
  // QKV projections (bf16 out)
  gemm_bf16<__bf16><<<dim3(2048 / 64, 4096 / 128), blk, 0, stream>>>(
      xb, Wqb, qtok, 4096, 2048, 2048);
  gemm_bf16<__bf16><<<dim3(512 / 64, 4096 / 128), blk, 0, stream>>>(
      xb, Wkb, ktok, 4096, 512, 2048);
  gemm_bf16<__bf16><<<dim3(512 / 64, 4096 / 128), blk, 0, stream>>>(
      xb, Wvb, vtok, 4096, 512, 2048);

  // RoPE(q,k) + head-major transpose; V transpose
  rope_kernel<<<(B_ * T_ * H_ * 32) / 256, 256, 0, stream>>>(qtok, qhd, H_);
  rope_kernel<<<(B_ * T_ * HKV_ * 32) / 256, 256, 0, stream>>>(ktok, khd, HKV_);
  v_transpose<<<(B_ * T_ * HKV_ * D_) / 256, 256, 0, stream>>>(vtok, vhd);

  // Causal GQA flash attention
  attn_kernel<<<dim3(B_ * H_, T_ / 64), dim3(128), 0, stream>>>(qhd, khd, vhd, attn);

  // Output projection (fp32 out)
  gemm_bf16<float><<<dim3(2048 / 64, 4096 / 128), blk, 0, stream>>>(
      attn, Wob, out, 4096, 2048, 2048);
}
